// GNNDecoder_20220706030176
// MI455X (gfx1250) — matrix operand order repaired and checked by reading, not compiled, not run on hardware
//
#include <hip/hip_runtime.h>

#define NN 100000
#define EE 1600000
#define WAVES 4   // waves per block (128 threads), keeps LDS < 64KB per block

typedef __bf16 bf16_t;
typedef bf16_t v16bf __attribute__((ext_vector_type(16)));
typedef float  v8f   __attribute__((ext_vector_type(8)));

union Frag { v16bf v; unsigned u[8]; };

__device__ __forceinline__ unsigned short f2bf_bits(float f) {
  unsigned u = __builtin_bit_cast(unsigned, f);
  unsigned r = u + 0x7FFFu + ((u >> 16) & 1u);   // round-to-nearest-even
  return (unsigned short)(r >> 16);
}
__device__ __forceinline__ unsigned pack2bf(float lo, float hi) {
  return (unsigned)f2bf_bits(lo) | ((unsigned)f2bf_bits(hi) << 16);
}
__device__ __forceinline__ v8f vzero() {
  v8f z = {0.f, 0.f, 0.f, 0.f, 0.f, 0.f, 0.f, 0.f};
  return z;
}

// A tile in LDS: 16 rows x APITCH u32 (packed bf16 K-pairs, row-major).
// B in LDS: fragment-major swizzle; per (kc,nt,lane) 8 contiguous u32
// so each B fragment is two ds_load_b128.
// Computes a 16x64 f32 tile as 4 N-tiles via v_wmma_f32_16x16x32_bf16.
template <int APITCH, int NKC>
__device__ __forceinline__ void gemm_tile(const unsigned* Aw, const unsigned* Bsw,
                                          int lane, v8f* acc) {
  const int m = lane & 15;
  const int half = lane >> 4;
#pragma unroll
  for (int kc = 0; kc < NKC; ++kc) {
    Frag a;
    // 16-bit A 16x32 layout -> two contiguous u32 quads per lane
#pragma unroll
    for (int j = 0; j < 4; ++j) {
      a.u[j]     = Aw[m * APITCH + kc * 16 + half * 4 + j];
      a.u[4 + j] = Aw[m * APITCH + kc * 16 + 8 + half * 4 + j];
    }
#pragma unroll
    for (int nt = 0; nt < 4; ++nt) {
      Frag b;
      const unsigned* bp = Bsw + (((kc * 4) + nt) * 32 + lane) * 8;
#pragma unroll
      for (int v = 0; v < 8; ++v) b.u[v] = bp[v];
      acc[nt] = __builtin_amdgcn_wmma_f32_16x16x32_bf16(
          false, a.v, false, b.v, (short)0, acc[nt], false, false);
    }
  }
}

// Stage f32 weights [K][64] -> LDS fragment-major swizzled packed bf16 pairs.
// dst[(((kc*4)+nt)*32 + lane)*8 + v] = pack(w[k][n], w[k+1][n])
//   k = kc*32 + (lane>>4)*8 + (v<4 ? 2v : 16+2(v-4)),  n = (lane&15) + nt*16
__device__ __forceinline__ void stage_wsw(unsigned* dst, const float* __restrict__ w,
                                          int K, int NKC, int tid, int nthreads) {
  const int total = NKC * 1024;           // NKC*4*32*8
  for (int idx = tid; idx < total; idx += nthreads) {
    const int v    = idx & 7;
    const int lane = (idx >> 3) & 31;
    const int nt   = (idx >> 8) & 3;
    const int kc   = idx >> 10;
    const int k = kc * 32 + (lane >> 4) * 8 + ((v < 4) ? 2 * v : 16 + 2 * (v - 4));
    const int n = (lane & 15) + nt * 16;
    const float f0 = (k < K) ? w[k * 64 + n] : 0.f;
    const float f1 = (k + 1 < K) ? w[(k + 1) * 64 + n] : 0.f;
    dst[idx] = pack2bf(f0, f1);
  }
}

// ---------- small helper kernels ----------

__global__ void zh_kernel(const float* __restrict__ z, const float* __restrict__ lw,
                          const float* __restrict__ lb, float* __restrict__ zh) {
  const int n = threadIdx.x;  // 64 threads
  float s = lb[n];
  for (int k = 0; k < 32; ++k) s += z[k] * lw[k * 64 + n];
  zh[n] = s;
}

__global__ void zero_kernel(float* __restrict__ p) {
  p[(size_t)blockIdx.x * 256 + threadIdx.x] = 0.f;
}

// per-edge: e0 = pos[col]-pos[row]; e = LN(relu(e0@w1+b1)@w2+b2)  (EDIM=2)
__global__ void edge_encode_kernel(const int* __restrict__ row, const int* __restrict__ col,
                                   const float* __restrict__ pos,
                                   const float* __restrict__ w1, const float* __restrict__ b1,
                                   const float* __restrict__ w2, const float* __restrict__ b2,
                                   const float* __restrict__ g, const float* __restrict__ bb,
                                   float* __restrict__ e) {
  const int i = blockIdx.x * 256 + threadIdx.x;
  if (i >= EE) return;
  const int r = row[i], c = col[i];
  const float2 pr = ((const float2*)pos)[r];
  const float2 pc = ((const float2*)pos)[c];
  const float dx = pc.x - pr.x, dy = pc.y - pr.y;
  float o0 = b2[0], o1 = b2[1];
  for (int n = 0; n < 64; ++n) {
    float h = dx * w1[n] + dy * w1[64 + n] + b1[n];
    h = fmaxf(h, 0.f);
    o0 += h * w2[n * 2];
    o1 += h * w2[n * 2 + 1];
  }
  const float mu = 0.5f * (o0 + o1);
  const float d0 = o0 - mu, d1 = o1 - mu;
  const float rs = rsqrtf(0.5f * (d0 * d0 + d1 * d1) + 1e-5f);
  e[i * 2]     = d0 * rs * g[0] + bb[0];
  e[i * 2 + 1] = d1 * rs * g[1] + bb[1];
}

// ---------- node encode: u = LN(MLP2(zh + pos@pos_w + pos_b)) ----------

__global__ __launch_bounds__(128) void node_encode_kernel(
    const float* __restrict__ zh, const float* __restrict__ pos,
    const float* __restrict__ pw, const float* __restrict__ pb,
    const float* __restrict__ w1, const float* __restrict__ b1,
    const float* __restrict__ w2, const float* __restrict__ b2,
    const float* __restrict__ lng, const float* __restrict__ lnb,
    float* __restrict__ u) {
  __shared__ unsigned W1p[2 * 1024];
  __shared__ unsigned W2p[2 * 1024];
  __shared__ unsigned Apk[WAVES * 16 * 32];
  __shared__ __attribute__((aligned(16))) unsigned short Hs[WAVES * 16 * 64];
  __shared__ float Ys[WAVES * 16 * 64];

  const int tid = threadIdx.x, lane = tid & 31, wave = tid >> 5;
  stage_wsw(W1p, w1, 64, 2, tid, 128);
  stage_wsw(W2p, w2, 64, 2, tid, 128);

  const int ntiles = NN / 16;
  int tile = blockIdx.x * WAVES + wave;
  const bool valid = tile < ntiles;
  if (!valid) tile = ntiles - 1;   // redundant compute, store guarded
  const long m0 = (long)tile * 16;

  unsigned* Aw = Apk + wave * (16 * 32);
  unsigned short* Hw = Hs + wave * (16 * 64);
  float* Yw = Ys + wave * (16 * 64);
  const int half = lane >> 4, nl = lane & 15;

  const float2 zh2 = ((const float2*)zh)[lane];
  const float2 pw0 = ((const float2*)pw)[lane];
  const float2 pw1 = ((const float2*)(pw + 64))[lane];
  const float2 pb2 = ((const float2*)pb)[lane];
  for (int i = 0; i < 16; ++i) {
    const float2 p = ((const float2*)pos)[m0 + i];
    const float f0 = zh2.x + p.x * pw0.x + p.y * pw1.x + pb2.x;
    const float f1 = zh2.y + p.x * pw0.y + p.y * pw1.y + pb2.y;
    Aw[i * 32 + lane] = pack2bf(f0, f1);
  }
  __syncthreads();

  v8f acc[4] = {vzero(), vzero(), vzero(), vzero()};
  gemm_tile<32, 2>(Aw, W1p, lane, acc);
#pragma unroll
  for (int nt = 0; nt < 4; ++nt) {
    const int n = nl + nt * 16;
    const float bb = b1[n];
#pragma unroll
    for (int v = 0; v < 8; ++v) {
      float h = acc[nt][v] + bb;
      h = h > 0.f ? h : 0.f;
      Hw[(v + 8 * half) * 64 + n] = f2bf_bits(h);
    }
  }
  __syncthreads();

  v8f acc2[4] = {vzero(), vzero(), vzero(), vzero()};
  gemm_tile<32, 2>(reinterpret_cast<const unsigned*>(Hw), W2p, lane, acc2);
#pragma unroll
  for (int nt = 0; nt < 4; ++nt) {
    const int n = nl + nt * 16;
    const float bb = b2[n];
#pragma unroll
    for (int v = 0; v < 8; ++v)
      Yw[(v + 8 * half) * 64 + n] = acc2[nt][v] + bb;
  }
  __syncthreads();

  if (valid && lane < 16) {  // one lane per node does LayerNorm
    const float* yr = Yw + lane * 64;
    float mu = 0.f;
    for (int n = 0; n < 64; ++n) mu += yr[n];
    mu *= (1.f / 64.f);
    float var = 0.f;
    for (int n = 0; n < 64; ++n) { const float d = yr[n] - mu; var += d * d; }
    var *= (1.f / 64.f);
    const float rs = rsqrtf(var + 1e-5f);
    float* ur = u + (m0 + lane) * 64;
    for (int n = 0; n < 64; ++n) ur[n] = (yr[n] - mu) * rs * lng[n] + lnb[n];
  }
}

// ---------- edge message MLP + scatter-sum (the hot kernel) ----------

__global__ __launch_bounds__(128) void msg_kernel(
    const float* __restrict__ u, const float* __restrict__ efeat,
    const int* __restrict__ row, const int* __restrict__ col,
    const float* __restrict__ w1, const float* __restrict__ b1,
    const float* __restrict__ w2, const float* __restrict__ b2,
    float* __restrict__ agg, int ngroups) {
  __shared__ unsigned W1p[5 * 1024];                // K=130 padded to 160
  __shared__ unsigned W2p[2 * 1024];
  __shared__ unsigned Apk[WAVES * 16 * 80];
  __shared__ __attribute__((aligned(16))) unsigned short Hs[WAVES * 16 * 64];

  const int tid = threadIdx.x, lane = tid & 31, wave = tid >> 5;
  stage_wsw(W1p, w1, 130, 5, tid, 128);
  stage_wsw(W2p, w2, 64, 2, tid, 128);

  unsigned* Aw = Apk + wave * (16 * 80);
  unsigned short* Hw = Hs + wave * (16 * 64);
  const int half = lane >> 4, nl = lane & 15;

  for (int g = blockIdx.x; g < ngroups; g += gridDim.x) {
    const long e0 = ((long)g * WAVES + wave) * 16;   // 16 edges per wave
    __syncthreads();                                  // weights visible / LDS reuse
    for (int i = 0; i < 16; ++i) {
      const int r = row[e0 + i];
      const int c = col[e0 + i];
      const float2 tu = ((const float2*)(u + (long)r * 64))[lane];
      const float2 tc = ((const float2*)(u + (long)c * 64))[lane];
      Aw[i * 80 + lane]      = pack2bf(tu.x, tu.y);   // K 0..63
      Aw[i * 80 + 32 + lane] = pack2bf(tc.x, tc.y);   // K 64..127
      if (lane < 16) {                                // K 128..159 (e + zero pad)
        unsigned pv = 0u;
        if (lane == 0) {
          const float2 te = ((const float2*)efeat)[e0 + i];
          pv = pack2bf(te.x, te.y);
        }
        Aw[i * 80 + 64 + lane] = pv;
      }
    }
    __syncthreads();

    v8f acc[4] = {vzero(), vzero(), vzero(), vzero()};
    gemm_tile<80, 5>(Aw, W1p, lane, acc);             // [16,160] @ [160,64]
#pragma unroll
    for (int nt = 0; nt < 4; ++nt) {
      const int n = nl + nt * 16;
      const float bb = b1[n];
#pragma unroll
      for (int v = 0; v < 8; ++v) {
        float h = acc[nt][v] + bb;
        h = h > 0.f ? h : 0.f;
        Hw[(v + 8 * half) * 64 + n] = f2bf_bits(h);
      }
    }
    __syncthreads();

    v8f acc2[4] = {vzero(), vzero(), vzero(), vzero()};
    gemm_tile<32, 2>(reinterpret_cast<const unsigned*>(Hw), W2p, lane, acc2);
#pragma unroll
    for (int v = 0; v < 8; ++v) {                     // scatter-sum to agg[row]
      const int r = row[e0 + v + 8 * half];
      float* aggr = agg + (long)r * 64;
#pragma unroll
      for (int nt = 0; nt < 4; ++nt) {
        const int n = nl + nt * 16;
        atomicAdd(&aggr[n], acc2[nt][v] + b2[n]);
      }
    }
  }
}

// ---------- node update: u += MLP2([u | agg]) ----------

__global__ __launch_bounds__(128) void node_update_kernel(
    float* __restrict__ u, const float* __restrict__ agg,
    const float* __restrict__ w1, const float* __restrict__ b1,
    const float* __restrict__ w2, const float* __restrict__ b2) {
  __shared__ unsigned W1p[4 * 1024];                 // K=128
  __shared__ unsigned W2p[2 * 1024];
  __shared__ unsigned Apk[WAVES * 16 * 64];
  __shared__ __attribute__((aligned(16))) unsigned short Hs[WAVES * 16 * 64];

  const int tid = threadIdx.x, lane = tid & 31, wave = tid >> 5;
  stage_wsw(W1p, w1, 128, 4, tid, 128);
  stage_wsw(W2p, w2, 64, 2, tid, 128);
  __syncthreads();

  const int ntiles = NN / 16;
  int tile = blockIdx.x * WAVES + wave;
  const bool valid = tile < ntiles;
  if (!valid) tile = ntiles - 1;
  const long m0 = (long)tile * 16;

  unsigned* Aw = Apk + wave * (16 * 64);
  unsigned short* Hw = Hs + wave * (16 * 64);
  const int half = lane >> 4, nl = lane & 15;

  for (int i = 0; i < 16; ++i) {
    const float2 tu = ((const float2*)(u + (m0 + i) * 64))[lane];
    const float2 ta = ((const float2*)(agg + (m0 + i) * 64))[lane];
    Aw[i * 64 + lane]      = pack2bf(tu.x, tu.y);     // K 0..63   (u)
    Aw[i * 64 + 32 + lane] = pack2bf(ta.x, ta.y);     // K 64..127 (agg)
  }
  __syncthreads();

  v8f acc[4] = {vzero(), vzero(), vzero(), vzero()};
  gemm_tile<64, 4>(Aw, W1p, lane, acc);
#pragma unroll
  for (int nt = 0; nt < 4; ++nt) {
    const int n = nl + nt * 16;
    const float bb = b1[n];
#pragma unroll
    for (int v = 0; v < 8; ++v) {
      float h = acc[nt][v] + bb;
      h = h > 0.f ? h : 0.f;
      Hw[(v + 8 * half) * 64 + n] = f2bf_bits(h);
    }
  }
  __syncthreads();

  v8f acc2[4] = {vzero(), vzero(), vzero(), vzero()};
  gemm_tile<32, 2>(reinterpret_cast<const unsigned*>(Hw), W2p, lane, acc2);
  if (valid) {
#pragma unroll
    for (int v = 0; v < 8; ++v) {
      const long node = m0 + v + 8 * half;
#pragma unroll
      for (int nt = 0; nt < 4; ++nt) {
        const int n = nl + nt * 16;
        u[node * 64 + n] += acc2[nt][v] + b2[n];      // residual, in place
      }
    }
  }
}

// ---------- decode: out = relu(u@w1+b1)@w2+b2  (OUT=3) ----------

__global__ __launch_bounds__(128) void decode_kernel(
    const float* __restrict__ u,
    const float* __restrict__ w1, const float* __restrict__ b1,
    const float* __restrict__ w2, const float* __restrict__ b2,
    float* __restrict__ out) {
  __shared__ unsigned W1p[2 * 1024];
  __shared__ unsigned Apk[WAVES * 16 * 32];
  __shared__ float Hf[WAVES * 16 * 64];

  const int tid = threadIdx.x, lane = tid & 31, wave = tid >> 5;
  stage_wsw(W1p, w1, 64, 2, tid, 128);
  __syncthreads();

  const int ntiles = NN / 16;
  int tile = blockIdx.x * WAVES + wave;
  const bool valid = tile < ntiles;
  if (!valid) tile = ntiles - 1;
  const long m0 = (long)tile * 16;

  unsigned* Aw = Apk + wave * (16 * 32);
  float* Hw = Hf + wave * (16 * 64);
  const int half = lane >> 4, nl = lane & 15;

  for (int i = 0; i < 16; ++i) {
    const float2 tu = ((const float2*)(u + (m0 + i) * 64))[lane];
    Aw[i * 32 + lane] = pack2bf(tu.x, tu.y);
  }
  __syncthreads();

  v8f acc[4] = {vzero(), vzero(), vzero(), vzero()};
  gemm_tile<32, 2>(Aw, W1p, lane, acc);
#pragma unroll
  for (int nt = 0; nt < 4; ++nt) {
    const int n = nl + nt * 16;
    const float bb = b1[n];
#pragma unroll
    for (int v = 0; v < 8; ++v)
      Hw[(v + 8 * half) * 64 + n] = fmaxf(acc[nt][v] + bb, 0.f);
  }
  __syncthreads();

  if (valid && lane < 16) {
    const float* hr = Hw + lane * 64;
    const long node = m0 + lane;
    for (int j = 0; j < 3; ++j) {
      float s = b2[j];
      for (int n = 0; n < 64; ++n) s += hr[n] * w2[n * 3 + j];
      out[node * 3 + j] = s;
    }
  }
}

// ---------- launch ----------

extern "C" void kernel_launch(void* const* d_in, const int* in_sizes, int n_in,
                              void* d_out, int out_size, void* d_ws, size_t ws_size,
                              hipStream_t stream) {
  (void)in_sizes; (void)n_in; (void)out_size; (void)ws_size;
  const float* z        = (const float*)d_in[0];
  const int*   eidx     = (const int*)d_in[1];
  const float* pos      = (const float*)d_in[2];
  const float* latent_w = (const float*)d_in[4];
  const float* latent_b = (const float*)d_in[5];
  const float* pos_w    = (const float*)d_in[6];
  const float* pos_b    = (const float*)d_in[7];
  const float* m0w1     = (const float*)d_in[8];
  const float* m0b1     = (const float*)d_in[9];
  const float* m0w2     = (const float*)d_in[10];
  const float* m0b2     = (const float*)d_in[11];
  const float* ln0g     = (const float*)d_in[12];
  const float* ln0b     = (const float*)d_in[13];
  const float* m1w1     = (const float*)d_in[14];
  const float* m1b1     = (const float*)d_in[15];
  const float* m1w2     = (const float*)d_in[16];
  const float* m1b2     = (const float*)d_in[17];
  const float* ln1g     = (const float*)d_in[18];
  const float* ln1b     = (const float*)d_in[19];
  const float* gmw1     = (const float*)d_in[20];
  const float* gmb1     = (const float*)d_in[21];
  const float* gmw2     = (const float*)d_in[22];
  const float* gmb2     = (const float*)d_in[23];
  const float* gnw1     = (const float*)d_in[24];
  const float* gnb1     = (const float*)d_in[25];
  const float* gnw2     = (const float*)d_in[26];
  const float* gnb2     = (const float*)d_in[27];
  const float* d2w1     = (const float*)d_in[28];
  const float* d2b1     = (const float*)d_in[29];
  const float* d2w2     = (const float*)d_in[30];
  const float* d2b2     = (const float*)d_in[31];

  const int* row = eidx;
  const int* col = eidx + EE;

  float* ws  = (float*)d_ws;
  float* zh  = ws;
  float* u   = ws + 256;
  float* agg = u + (long)NN * 64;
  float* ef  = agg + (long)NN * 64;

  const int node_blocks = (NN / 16 + WAVES - 1) / WAVES;  // 1563

  zh_kernel<<<1, 64, 0, stream>>>(z, latent_w, latent_b, zh);
  edge_encode_kernel<<<EE / 256, 256, 0, stream>>>(row, col, pos, m1w1, m1b1,
                                                   m1w2, m1b2, ln1g, ln1b, ef);
  node_encode_kernel<<<node_blocks, 128, 0, stream>>>(zh, pos, pos_w, pos_b,
                                                      m0w1, m0b1, m0w2, m0b2,
                                                      ln0g, ln0b, u);
  for (int l = 0; l < 3; ++l) {
    zero_kernel<<<(NN * 64) / 256, 256, 0, stream>>>(agg);
    msg_kernel<<<2500, 128, 0, stream>>>(u, ef, row, col,
        gmw1 + (long)l * 130 * 64, gmb1 + l * 64,
        gmw2 + (long)l * 64 * 64, gmb2 + l * 64, agg, EE / 64);
    node_update_kernel<<<node_blocks, 128, 0, stream>>>(u, agg,
        gnw1 + (long)l * 128 * 64, gnb1 + l * 64,
        gnw2 + (long)l * 64 * 64, gnb2 + l * 64);
  }
  decode_kernel<<<node_blocks, 128, 0, stream>>>(u, d2w1, d2b1, d2w2, d2b2,
                                                 (float*)d_out);
}
